// SFGTransformer_52055003627703
// MI455X (gfx1250) — matrix operand order repaired and checked by reading, not compiled, not run on hardware
//
#include <hip/hip_runtime.h>
#include <hip/hip_fp16.h>
#include <math.h>
#include <stdint.h>

#define NND 50000
#define NE  800000
#define FIN 32
#define DM  128
#define NL  4
#define NH  8
#define DKH 16
#define NB  5
#define NS  5
#define DFF 512

typedef __attribute__((ext_vector_type(16))) _Float16 v16h;
typedef __attribute__((ext_vector_type(4)))  _Float16 v4h;
typedef __attribute__((ext_vector_type(8)))  float    v8f;
typedef __attribute__((ext_vector_type(4)))  unsigned int u32x4;
typedef __attribute__((ext_vector_type(4)))  int      i32x4;
typedef __attribute__((ext_vector_type(8)))  int      i32x8;

// ---- WMMA A-fragment index map (contiguous k = 16*half + e, matches packed B) ----
__device__ __forceinline__ int kmapA(int e, int half) {
  return half * 16 + e;
}

__device__ __forceinline__ v16h load_fragA_lds(const _Float16* A, int r, int k0,
                                               int stride, int half) {
  v16h a;
#pragma unroll
  for (int e = 0; e < 16; ++e) a[e] = A[r * stride + k0 + kmapA(e, half)];
  return a;
}
// B pre-swizzled: tile = ct*nkt + kt, each lane's 16 halves contiguous (32B aligned)
__device__ __forceinline__ v16h load_fragB_pk(const _Float16* P, int tile, int lane) {
  return *(const v16h*)(P + ((size_t)tile << 9) + (lane << 4));
}
__device__ __forceinline__ v8f wmma(v16h a, v16h b, v8f c) {
  return __builtin_amdgcn_wmma_f32_16x16x32_f16(false, a, false, b, (short)0, c,
                                                false, false);
}

// ---- TDM: DMA a 16x128 f16 tile (4KB, contiguous rows) from global into LDS ----
// D# per CDNA5 ISA ch.8: group0 {count=1, lds_addr, global_addr, type=2},
// group1 {data_size=2B, tensor 128x16, tile 128x16, dim0_stride=128}, groups2/3 = 0.
__device__ __forceinline__ void tdm_load_tile_16x128_f16(const _Float16* gsrc,
                                                         _Float16* lds) {
  unsigned laddr = (unsigned)(uintptr_t)lds;  // LDS byte offset (low 32b of flat addr)
  unsigned long long ga = (unsigned long long)(uintptr_t)gsrc;
  u32x4 g0;
  g0[0] = 1u;                                       // count = 1 valid descriptor
  g0[1] = laddr;                                    // lds_addr
  g0[2] = (unsigned)(ga & 0xFFFFFFFFu);             // global_addr[31:0]
  g0[3] = (unsigned)((ga >> 32) & 0x01FFFFFFu)      // global_addr[56:32]
          | (2u << 30);                             // type = 2 ("image")
  i32x8 g1;
  g1[0] = 0x00010000;         // workgroup_mask=0, data_size=1 (2 bytes/elem)
  g1[1] = (int)(128u << 16);  // tensor_dim0 = 128 (low 16 bits at bit48)
  g1[2] = (int)(16u << 16);   // tensor_dim0 hi = 0 | tensor_dim1 = 16
  g1[3] = (int)(128u << 16);  // tensor_dim1 hi = 0 | tile_dim0 = 128
  g1[4] = 16;                 // tile_dim1 = 16, tile_dim2 = 0
  g1[5] = 128;                // tensor_dim0_stride = 128
  g1[6] = 0;
  g1[7] = 0;
  i32x4 z4 = {0, 0, 0, 0};
#if __clang_major__ >= 23
  i32x8 z8 = {0, 0, 0, 0, 0, 0, 0, 0};
  __builtin_amdgcn_tensor_load_to_lds(g0, g1, z4, z4, z8, 0);
#else
  __builtin_amdgcn_tensor_load_to_lds(g0, g1, z4, z4, 0);
#endif
}

// ---- monotone float<->uint encoding for atomic max on signed floats ----
__device__ __forceinline__ unsigned fenc(float f) {
  unsigned u = __float_as_uint(f);
  return (u & 0x80000000u) ? ~u : (u | 0x80000000u);
}
__device__ __forceinline__ float fdec(unsigned e) {
  unsigned u = (e & 0x80000000u) ? (e & 0x7FFFFFFFu) : ~e;
  return __uint_as_float(u);
}

// ---- weight pack: fp32 (b,K,N) -> f16 fragment-contiguous layout ----
__global__ void pack_w_kernel(const float* __restrict__ src, _Float16* __restrict__ dst,
                              int K, int N, int total) {
  int per = K * N;
  int nkt = K >> 5;
  for (int idx = blockIdx.x * blockDim.x + threadIdx.x; idx < total;
       idx += gridDim.x * blockDim.x) {
    int b = idx / per, rem = idx - b * per;
    int e = rem & 15;
    int lane = (rem >> 4) & 31;
    int t = rem >> 9;
    int kt = t % nkt, ct = t / nkt;
    int half = lane >> 4, col = lane & 15;
    int sk = kt * 32 + half * 16 + e;
    int sn = ct * 16 + col;
    dst[idx] = (_Float16)src[(size_t)b * per + (size_t)sk * N + sn];
  }
}

// ---- per-layer scratch init ----
__global__ void init_layer_kernel(unsigned* __restrict__ nodemax,
                                  float* __restrict__ nodesum,
                                  float* __restrict__ agg) {
  int stride = gridDim.x * blockDim.x;
  int i0 = blockIdx.x * blockDim.x + threadIdx.x;
  const unsigned NEG_INF = 0x007FFFFFu;  // fenc(-inf)
  for (int j = i0; j < NND * NH; j += stride) { nodemax[j] = NEG_INF; nodesum[j] = 0.f; }
  for (int j = i0; j < NND * DM; j += stride) agg[j] = 0.f;
}

// ---- stage a 16xW fp32 tile into LDS as f16 via float4 loads ----
__device__ __forceinline__ void stage_tile(const float* __restrict__ src,
                                           _Float16* __restrict__ At,
                                           int nfloat4, int t) {
  const float4* s4 = (const float4*)src;
  for (int i = t; i < nfloat4; i += 256) {
    float4 v = s4[i];
    v4h hv = {(_Float16)v.x, (_Float16)v.y, (_Float16)v.z, (_Float16)v.w};
    *(v4h*)&At[i * 4] = hv;
  }
}

// ---------------- h = x @ in_w + in_b + stage_emb[stage_ids] ----------------
__global__ __launch_bounds__(256) void input_proj_kernel(
    const float* __restrict__ x, const _Float16* __restrict__ Wp,
    const float* __restrict__ in_b, const float* __restrict__ stage_emb,
    const int* __restrict__ stage_ids, float* __restrict__ h,
    _Float16* __restrict__ h16) {
  __shared__ _Float16 At[16 * FIN];
  int row0 = blockIdx.x * 16, t = threadIdx.x;
  stage_tile(&x[(size_t)row0 * FIN], At, 16 * FIN / 4, t);
  __syncthreads();
  int wave = t >> 5, lane = t & 31, r = lane & 15, half = lane >> 4;
  v8f c = {};
  c = wmma(load_fragA_lds(At, r, 0, FIN, half), load_fragB_pk(Wp, wave, lane), c);
#pragma unroll
  for (int i = 0; i < 8; ++i) {
    int m = i + 8 * half, n = wave * 16 + r, row = row0 + m;
    float v = c[i] + in_b[n] + stage_emb[stage_ids[row] * DM + n];
    h[(size_t)row * DM + n] = v;
    h16[(size_t)row * DM + n] = (_Float16)v;
  }
}

// ---------------- Q/K/V = h @ W{q,k,v}, stored as f16 ----------------
__global__ __launch_bounds__(256) void qkv_kernel(
    const _Float16* __restrict__ h16, const _Float16* __restrict__ Wq,
    const _Float16* __restrict__ Wk, const _Float16* __restrict__ Wv,
    _Float16* __restrict__ Qh, _Float16* __restrict__ Kh, _Float16* __restrict__ Vh) {
  __shared__ _Float16 At[16 * DM];
  int row0 = blockIdx.x * 16, t = threadIdx.x;
  if (t < 32) {  // one wave issues the DMA; TDM ignores EXEC
    tdm_load_tile_16x128_f16(h16 + (size_t)row0 * DM, At);
    __builtin_amdgcn_s_wait_tensorcnt(0);
  }
  __syncthreads();
  int wave = t >> 5, lane = t & 31, r = lane & 15, half = lane >> 4;
  v16h af[4];
#pragma unroll
  for (int kt = 0; kt < 4; ++kt) af[kt] = load_fragA_lds(At, r, kt * 32, DM, half);
  const _Float16* Bs[3] = {Wq, Wk, Wv};
  _Float16* Os[3] = {Qh, Kh, Vh};
#pragma unroll
  for (int j = 0; j < 3; ++j) {
    v8f c = {};
#pragma unroll
    for (int kt = 0; kt < 4; ++kt)
      c = wmma(af[kt], load_fragB_pk(Bs[j], wave * 4 + kt, lane), c);
#pragma unroll
    for (int i = 0; i < 8; ++i) {
      int m = i + 8 * half;
      Os[j][(size_t)(row0 + m) * DM + wave * 16 + r] = (_Float16)c[i];
    }
  }
}

// ---------------- edge attention logits + segment max ----------------
__global__ void edge_logits_kernel(
    const _Float16* __restrict__ Qh, const _Float16* __restrict__ Kh,
    const int* __restrict__ src, const int* __restrict__ dst,
    const float* __restrict__ edge_attr, const int* __restrict__ band_ids,
    const float* __restrict__ bb, float* __restrict__ logits,
    unsigned* __restrict__ nodemax) {
  int idx = blockIdx.x * blockDim.x + threadIdx.x;
  if (idx >= NE * NH) return;
  int e = idx >> 3, hh = idx & 7;
  int s = src[e], d = dst[e];
  v16h qv = *(const v16h*)(Qh + (size_t)d * DM + hh * DKH);
  v16h kv = *(const v16h*)(Kh + (size_t)s * DM + hh * DKH);
  float acc = 0.f;
#pragma unroll
  for (int i = 0; i < DKH; ++i) acc += (float)qv[i] * (float)kv[i];
  float lg = acc * 0.25f;  // 1/sqrt(16)
  lg = (lg + bb[band_ids[e] * NH + hh]) * edge_attr[e];
  logits[idx] = lg;
  atomicMax(&nodemax[d * NH + hh], fenc(lg));
}

// ---------------- exp(logit - max) + segment sum ----------------
__global__ void edge_exp_kernel(const int* __restrict__ dst,
                                float* __restrict__ logits,
                                const unsigned* __restrict__ nodemax,
                                float* __restrict__ nodesum) {
  int idx = blockIdx.x * blockDim.x + threadIdx.x;
  if (idx >= NE * NH) return;
  int e = idx >> 3, hh = idx & 7;
  int d = dst[e];
  float ex = __expf(logits[idx] - fdec(nodemax[d * NH + hh]));
  logits[idx] = ex;
  atomicAdd(&nodesum[d * NH + hh], ex);
}

// ---------------- normalized weighted scatter of V ----------------
__global__ void edge_agg_kernel(const int* __restrict__ src, const int* __restrict__ dst,
                                const float* __restrict__ logits,
                                const float* __restrict__ nodesum,
                                const _Float16* __restrict__ Vh,
                                float* __restrict__ agg) {
  int idx = blockIdx.x * blockDim.x + threadIdx.x;
  if (idx >= NE * NH) return;
  int e = idx >> 3, hh = idx & 7;
  int s = src[e], d = dst[e];
  float a = logits[idx] / (nodesum[d * NH + hh] + 1e-16f);
  v16h vv = *(const v16h*)(Vh + (size_t)s * DM + hh * DKH);
  float* o = agg + (size_t)d * DM + hh * DKH;
#pragma unroll
  for (int i = 0; i < DKH; ++i) atomicAdd(&o[i], a * (float)vv[i]);
}

// ---------------- row-pair layernorm; writes f32 h and f16 mirror ----------------
__device__ __forceinline__ void ln_rows(const float* Ot, int row0, int wave, int lane,
                                        const float* g, const float* b,
                                        float* __restrict__ h,
                                        _Float16* __restrict__ h16) {
  for (int rr = wave * 2; rr < wave * 2 + 2; ++rr) {
    float s = 0.f;
    for (int j = lane; j < DM; j += 32) s += Ot[rr * DM + j];
#pragma unroll
    for (int o = 16; o; o >>= 1) s += __shfl_xor(s, o);
    float mean = s * (1.0f / DM);
    float v = 0.f;
    for (int j = lane; j < DM; j += 32) { float d = Ot[rr * DM + j] - mean; v += d * d; }
#pragma unroll
    for (int o = 16; o; o >>= 1) v += __shfl_xor(v, o);
    float rs = rsqrtf(v * (1.0f / DM) + 1e-5f);
    for (int j = lane; j < DM; j += 32) {
      float out = (Ot[rr * DM + j] - mean) * rs * g[j] + b[j];
      h[(size_t)(row0 + rr) * DM + j] = out;
      h16[(size_t)(row0 + rr) * DM + j] = (_Float16)out;
    }
  }
}

// ---------------- o = agg @ Wo + b; h = LN1(o + h) ----------------
__global__ __launch_bounds__(256) void attn_out_kernel(
    const float* __restrict__ agg, const _Float16* __restrict__ Wo,
    const float* __restrict__ Wo_b, const float* __restrict__ ln_g,
    const float* __restrict__ ln_b, float* __restrict__ h,
    _Float16* __restrict__ h16) {
  __shared__ _Float16 At[16 * DM];
  __shared__ float Ot[16 * DM];
  int row0 = blockIdx.x * 16, t = threadIdx.x;
  stage_tile(&agg[(size_t)row0 * DM], At, 16 * DM / 4, t);
  __syncthreads();
  int wave = t >> 5, lane = t & 31, r = lane & 15, half = lane >> 4;
  v8f c = {};
#pragma unroll
  for (int kt = 0; kt < 4; ++kt)
    c = wmma(load_fragA_lds(At, r, kt * 32, DM, half),
             load_fragB_pk(Wo, wave * 4 + kt, lane), c);
#pragma unroll
  for (int i = 0; i < 8; ++i) {
    int m = i + 8 * half, n = wave * 16 + r;
    Ot[m * DM + n] = c[i] + Wo_b[n] + h[(size_t)(row0 + m) * DM + n];
  }
  __syncthreads();
  ln_rows(Ot, row0, wave, lane, ln_g, ln_b, h, h16);
}

// ---------------- fused FFN: gelu(h@W1+b1)@W2+b2 residual + LN2 ----------------
__global__ __launch_bounds__(256) void ffn_kernel(
    const _Float16* __restrict__ W1, const float* __restrict__ b1,
    const _Float16* __restrict__ W2, const float* __restrict__ b2,
    const float* __restrict__ ln_g, const float* __restrict__ ln_b,
    float* __restrict__ h, _Float16* __restrict__ h16) {
  __shared__ _Float16 At[16 * DM];
  __shared__ _Float16 Hid[16 * DFF];
  __shared__ float Ot[16 * DM];
  int row0 = blockIdx.x * 16, t = threadIdx.x;
  if (t < 32) {
    tdm_load_tile_16x128_f16(h16 + (size_t)row0 * DM, At);
    __builtin_amdgcn_s_wait_tensorcnt(0);
  }
  __syncthreads();
  int wave = t >> 5, lane = t & 31, r = lane & 15, half = lane >> 4;
  v16h af[4];
#pragma unroll
  for (int kt = 0; kt < 4; ++kt) af[kt] = load_fragA_lds(At, r, kt * 32, DM, half);
  // GEMM1 + exact GeLU: 32 col tiles of DFF across 8 waves (nkt = 4)
#pragma unroll
  for (int tt = 0; tt < 4; ++tt) {
    int ct = wave + tt * 8;
    v8f c = {};
#pragma unroll
    for (int kt = 0; kt < 4; ++kt)
      c = wmma(af[kt], load_fragB_pk(W1, ct * 4 + kt, lane), c);
#pragma unroll
    for (int i = 0; i < 8; ++i) {
      int m = i + 8 * half, n = ct * 16 + r;
      float xg = c[i] + b1[n];
      Hid[m * DFF + n] = (_Float16)(0.5f * xg * (1.0f + erff(xg * 0.70710678118f)));
    }
  }
  __syncthreads();
  // GEMM2: 8 output col tiles, K = 512 (nkt = 16)
  {
    v8f c = {};
#pragma unroll
    for (int kt = 0; kt < 16; ++kt)
      c = wmma(load_fragA_lds(Hid, r, kt * 32, DFF, half),
               load_fragB_pk(W2, wave * 16 + kt, lane), c);
#pragma unroll
    for (int i = 0; i < 8; ++i) {
      int m = i + 8 * half, n = wave * 16 + r;
      Ot[m * DM + n] = c[i] + b2[n] + h[(size_t)(row0 + m) * DM + n];
    }
  }
  __syncthreads();
  ln_rows(Ot, row0, wave, lane, ln_g, ln_b, h, h16);
}

// ---------------- out = h @ out_w + out_b ----------------
__global__ __launch_bounds__(256) void out_proj_kernel(
    const _Float16* __restrict__ h16, const _Float16* __restrict__ Wp,
    const float* __restrict__ ob, float* __restrict__ out) {
  __shared__ _Float16 At[16 * DM];
  int row0 = blockIdx.x * 16, t = threadIdx.x;
  if (t < 32) {
    tdm_load_tile_16x128_f16(h16 + (size_t)row0 * DM, At);
    __builtin_amdgcn_s_wait_tensorcnt(0);
  }
  __syncthreads();
  int wave = t >> 5, lane = t & 31, r = lane & 15, half = lane >> 4;
  v8f c = {};
#pragma unroll
  for (int kt = 0; kt < 4; ++kt)
    c = wmma(load_fragA_lds(At, r, kt * 32, DM, half),
             load_fragB_pk(Wp, wave * 4 + kt, lane), c);
#pragma unroll
  for (int i = 0; i < 8; ++i) {
    int m = i + 8 * half, n = wave * 16 + r;
    out[(size_t)(row0 + m) * DM + n] = c[i] + ob[n];
  }
}

extern "C" void kernel_launch(void* const* d_in, const int* in_sizes, int n_in,
                              void* d_out, int out_size, void* d_ws, size_t ws_size,
                              hipStream_t stream) {
  const float* x          = (const float*)d_in[0];
  const int*   edge_index = (const int*)d_in[1];
  const int*   src        = edge_index;
  const int*   dst        = edge_index + NE;
  const float* edge_attr  = (const float*)d_in[2];
  const int*   band_ids   = (const int*)d_in[3];
  const int*   stage_ids  = (const int*)d_in[4];
  const float* stage_emb  = (const float*)d_in[5];
  const float* in_w       = (const float*)d_in[6];
  const float* in_b       = (const float*)d_in[7];
  const float* Wq         = (const float*)d_in[8];
  const float* Wk         = (const float*)d_in[9];
  const float* Wv         = (const float*)d_in[10];
  const float* Wo         = (const float*)d_in[11];
  const float* Wo_b       = (const float*)d_in[12];
  const float* band_bias  = (const float*)d_in[13];
  const float* ln1_g      = (const float*)d_in[14];
  const float* ln1_b      = (const float*)d_in[15];
  const float* ffn_w1     = (const float*)d_in[16];
  const float* ffn_b1     = (const float*)d_in[17];
  const float* ffn_w2     = (const float*)d_in[18];
  const float* ffn_b2     = (const float*)d_in[19];
  const float* ln2_g      = (const float*)d_in[20];
  const float* ln2_b      = (const float*)d_in[21];
  const float* out_w      = (const float*)d_in[22];
  const float* out_b      = (const float*)d_in[23];

  // ---- workspace carve ----
  char* p = (char*)d_ws;
  auto alloc = [&](size_t bytes) {
    void* r = (void*)p;
    p += (bytes + 255) & ~(size_t)255;
    return r;
  };
  float*     h        = (float*)alloc((size_t)NND * DM * 4);
  _Float16*  h16      = (_Float16*)alloc((size_t)NND * DM * 2);
  float*     aggb     = (float*)alloc((size_t)NND * DM * 4);
  float*     logits   = (float*)alloc((size_t)NE * NH * 4);
  unsigned*  nodemax  = (unsigned*)alloc((size_t)NND * NH * 4);
  float*     nodesum  = (float*)alloc((size_t)NND * NH * 4);
  _Float16*  Qh       = (_Float16*)alloc((size_t)NND * DM * 2);
  _Float16*  Kh       = (_Float16*)alloc((size_t)NND * DM * 2);
  _Float16*  Vh       = (_Float16*)alloc((size_t)NND * DM * 2);
  _Float16*  in_w_p   = (_Float16*)alloc((size_t)FIN * DM * 2);
  _Float16*  Wq_p     = (_Float16*)alloc((size_t)NL * DM * DM * 2);
  _Float16*  Wk_p     = (_Float16*)alloc((size_t)NL * DM * DM * 2);
  _Float16*  Wv_p     = (_Float16*)alloc((size_t)NL * DM * DM * 2);
  _Float16*  Wo_p     = (_Float16*)alloc((size_t)NL * DM * DM * 2);
  _Float16*  W1_p     = (_Float16*)alloc((size_t)NL * DM * DFF * 2);
  _Float16*  W2_p     = (_Float16*)alloc((size_t)NL * DFF * DM * 2);
  _Float16*  out_w_p  = (_Float16*)alloc((size_t)DM * DM * 2);

  auto pack = [&](const float* s, _Float16* d, int K, int N, int batch) {
    int total = batch * K * N;
    int blocks = (total + 255) / 256;
    if (blocks > 2048) blocks = 2048;
    pack_w_kernel<<<blocks, 256, 0, stream>>>(s, d, K, N, total);
  };
  pack(in_w,   in_w_p,  FIN, DM,  1);
  pack(Wq,     Wq_p,    DM,  DM,  NL);
  pack(Wk,     Wk_p,    DM,  DM,  NL);
  pack(Wv,     Wv_p,    DM,  DM,  NL);
  pack(Wo,     Wo_p,    DM,  DM,  NL);
  pack(ffn_w1, W1_p,    DM,  DFF, NL);
  pack(ffn_w2, W2_p,    DFF, DM,  NL);
  pack(out_w,  out_w_p, DM,  DM,  1);

  const int RB = NND / 16;  // 3125 row blocks (exact)
  input_proj_kernel<<<RB, 256, 0, stream>>>(x, in_w_p, in_b, stage_emb, stage_ids,
                                            h, h16);

  const int ETH = NE * NH;
  const int EB  = (ETH + 255) / 256;

  for (int l = 0; l < NL; ++l) {
    init_layer_kernel<<<1024, 256, 0, stream>>>(nodemax, nodesum, aggb);
    qkv_kernel<<<RB, 256, 0, stream>>>(h16, Wq_p + (size_t)l * DM * DM,
                                       Wk_p + (size_t)l * DM * DM,
                                       Wv_p + (size_t)l * DM * DM, Qh, Kh, Vh);
    edge_logits_kernel<<<EB, 256, 0, stream>>>(Qh, Kh, src, dst, edge_attr, band_ids,
                                               band_bias + (size_t)l * NB * NH,
                                               logits, nodemax);
    edge_exp_kernel<<<EB, 256, 0, stream>>>(dst, logits, nodemax, nodesum);
    edge_agg_kernel<<<EB, 256, 0, stream>>>(src, dst, logits, nodesum, Vh, aggb);
    attn_out_kernel<<<RB, 256, 0, stream>>>(aggb, Wo_p + (size_t)l * DM * DM,
                                            Wo_b + (size_t)l * DM,
                                            ln1_g + (size_t)l * DM,
                                            ln1_b + (size_t)l * DM, h, h16);
    ffn_kernel<<<RB, 256, 0, stream>>>(W1_p + (size_t)l * DM * DFF,
                                       ffn_b1 + (size_t)l * DFF,
                                       W2_p + (size_t)l * DFF * DM,
                                       ffn_b2 + (size_t)l * DM,
                                       ln2_g + (size_t)l * DM,
                                       ln2_b + (size_t)l * DM, h, h16);
  }
  out_proj_kernel<<<RB, 256, 0, stream>>>(h16, out_w_p, out_b, (float*)d_out);
}
